// Slot_Generator_89696097010163
// MI455X (gfx1250) — matrix operand order repaired and checked by reading, not compiled, not run on hardware
//
#include <hip/hip_runtime.h>
#include <hip/hip_bf16.h>
#include <math.h>

// ---------------------------------------------------------------------------
// Problem constants (from reference setup_inputs)
// ---------------------------------------------------------------------------
static constexpr int BN  = 8;      // batch
static constexpr int CIN = 512;    // input channels
static constexpr int CO  = 128;    // squeezed channels (Cout)
static constexpr int SN  = 16;     // slot channels
static constexpr int HW  = 4096;   // H*W (64x64)
static constexpr int RR  = 3;      // refs
static constexpr int RB  = RR * BN;      // 24
static constexpr int KC  = 64;     // kmeans clusters
static constexpr int KV  = RR * CO;      // 384 kv tokens
static constexpr int KM_ITERS = 10;

#ifndef __has_builtin
#define __has_builtin(x) 0
#endif
#if __has_builtin(__builtin_amdgcn_global_load_async_to_lds_b128) && \
    __has_builtin(__builtin_amdgcn_s_wait_asynccnt)
#define USE_ASYNC_LDS 1
#else
#define USE_ASYNC_LDS 0
#endif

// ---------------------------------------------------------------------------
// WMMA plumbing (CDNA5 wave32, v_wmma_f32_16x16x32_bf16)
// ---------------------------------------------------------------------------
typedef __attribute__((ext_vector_type(16))) __bf16 v16bf;
typedef __attribute__((ext_vector_type(8)))  float  v8f;
typedef __attribute__((ext_vector_type(4)))  int    v4i;
typedef __attribute__((address_space(1))) v4i* as1_v4i_p;
typedef __attribute__((address_space(3))) v4i* as3_v4i_p;

__device__ __forceinline__ v8f wmma_bf16(v16bf a, v16bf b, v8f c) {
  return __builtin_amdgcn_wmma_f32_16x16x32_bf16(
      /*neg_a=*/false, a, /*neg_b=*/false, b,
      /*c_mod=*/(short)0, c, /*reuse_a=*/false, /*reuse_b=*/false);
}

// A-matrix 16x32 bf16 fragment, A row-major (k contiguous), row folded in ptr.
// VGPR i pair covers K = 16*half + 2i.
__device__ __forceinline__ v16bf load_a_rm(const float* __restrict__ arow,
                                           int kk, int half) {
  const float* ar = arow + kk + half * 16;
  v16bf a;
#pragma unroll
  for (int i = 0; i < 8; ++i) {
    int kb = i << 1; // 2i
    a[2 * i]     = (__bf16)ar[kb];
    a[2 * i + 1] = (__bf16)ar[kb + 1];
  }
  return a;
}

// Same, with fused softmax on the A row: a' = exp(a - mx) * rinv
__device__ __forceinline__ v16bf load_a_rm_soft(const float* __restrict__ arow,
                                                int kk, int half,
                                                float mx, float rinv) {
  const float* ar = arow + kk + half * 16;
  v16bf a;
#pragma unroll
  for (int i = 0; i < 8; ++i) {
    int kb = i << 1;
    float x0 = __expf(ar[kb] - mx) * rinv;
    float x1 = __expf(ar[kb + 1] - mx) * rinv;
    a[2 * i]     = (__bf16)x0;
    a[2 * i + 1] = (__bf16)x1;
  }
  return a;
}

// B-matrix 32x16 bf16 fragment, B stored K x N (ldb = N). Lane col = n.
// VGPR i pair covers K = 16*half + 2i.
__device__ __forceinline__ v16bf load_b_kn(const float* __restrict__ B,
                                           int ldb, int kk, int n, int half) {
  const float* bp = B + (size_t)(kk + half * 16) * ldb + n;
  v16bf b;
#pragma unroll
  for (int i = 0; i < 8; ++i) {
    b[2 * i]     = (__bf16)bp[(size_t)(2 * i) * ldb];
    b[2 * i + 1] = (__bf16)bp[(size_t)(2 * i + 1) * ldb];
  }
  return b;
}

// B^T fragment: B stored N x K row-major (ldb = K); pairs are contiguous.
__device__ __forceinline__ v16bf load_b_nk(const float* __restrict__ brow,
                                           int kk, int half) {
  const float* bp = brow + kk + half * 16;
  v16bf b;
#pragma unroll
  for (int i = 0; i < 8; ++i) {
    b[2 * i]     = (__bf16)bp[2 * i];
    b[2 * i + 1] = (__bf16)bp[2 * i + 1];
  }
  return b;
}

// B fragment from an LDS-staged 32(K) x 128(N) fp32 tile, row stride 132.
__device__ __forceinline__ v16bf load_b_lds(const float* __restrict__ lbuf,
                                            int n, int half) {
  const float* bp = lbuf + (half * 16) * 132 + n;
  v16bf b;
#pragma unroll
  for (int i = 0; i < 8; ++i) {
    b[2 * i]     = (__bf16)bp[(2 * i) * 132];
    b[2 * i + 1] = (__bf16)bp[(2 * i + 1) * 132];
  }
  return b;
}

// ---------------------------------------------------------------------------
// Bulk conv GEMM: Y[z](CO x HW) = W(CO x K) @ X[z](K x HW) + bias
// Block = 256 threads (8 waves), tile 128(M) x 128(N); the 32x128 fp32
// activation sub-tile is staged in LDS (async-to-LDS with double buffering
// when available) and shared by all 8 waves -> activations read once from HBM.
// ---------------------------------------------------------------------------
__device__ __forceinline__ void stage_tile(const float* __restrict__ Xk,
                                           float* __restrict__ lbuf, int t) {
#pragma unroll
  for (int j = 0; j < 4; ++j) {
    int e = t * 4 + j * 1024;       // 4096 floats / 256 threads, 4 per shot
    int k = e >> 7, n = e & 127;
    const float* g = Xk + (size_t)k * HW + n;
    float* l = lbuf + k * 132 + n;  // 16B-aligned (n%4==0, 132%4==0)
#if USE_ASYNC_LDS
    __builtin_amdgcn_global_load_async_to_lds_b128(
        (as1_v4i_p)(uintptr_t)g, (as3_v4i_p)l, 0, 0);
#else
    *(float4*)l = *(const float4*)g;
#endif
  }
}

__global__ void __launch_bounds__(256)
conv_gemm_kernel(const float* __restrict__ W, const float* __restrict__ bias,
                 const float* __restrict__ X, long xStr,
                 float* __restrict__ Y, long yStr, int K) {
  __shared__ float tile[2][32 * 132];
  const int t = threadIdx.x;
  const int lane = t & 31, wave = t >> 5;
  const int half = lane >> 4, nl = lane & 15;
  const int n0 = blockIdx.x * 128;
  const int z  = blockIdx.z;
  const float* Xz   = X + (size_t)z * xStr + n0;
  const float* Arow = W + (size_t)(wave * 16 + nl) * K;
  v8f acc[8] = {};
  const int nsteps = K / 32;
  stage_tile(Xz, tile[0], t);
  for (int s = 0; s < nsteps; ++s) {
    if (s + 1 < nsteps) {
      stage_tile(Xz + (size_t)(s + 1) * 32 * HW, tile[(s + 1) & 1], t);
#if USE_ASYNC_LDS
      __builtin_amdgcn_s_wait_asynccnt(4);   // retire current stage (4 newest pending)
#endif
    } else {
#if USE_ASYNC_LDS
      __builtin_amdgcn_s_wait_asynccnt(0);
#endif
    }
    __syncthreads();
    const float* lb = tile[s & 1];
    v16bf a = load_a_rm(Arow, s * 32, half);
#pragma unroll
    for (int tt = 0; tt < 8; ++tt) {
      v16bf b = load_b_lds(lb, tt * 16 + nl, half);
      acc[tt] = wmma_bf16(a, b, acc[tt]);
    }
    __syncthreads();
  }
#pragma unroll
  for (int tt = 0; tt < 8; ++tt) {
    int n = n0 + tt * 16 + nl;
#pragma unroll
    for (int r = 0; r < 8; ++r) {
      int m = wave * 16 + half * 8 + r;
      Y[(size_t)z * yStr + (size_t)m * HW + n] = acc[tt][r] + bias[m];
    }
  }
}

// ---------------------------------------------------------------------------
// Generic GEMM kernels: one wave (32 threads) per 16(M) x 64(N) output strip.
// grid = (N/64, M/16, batch). Requires M%16==0, N%64==0, K%32==0.
// ---------------------------------------------------------------------------

// Y[z] = A[z](MxK, row-major) @ B[z](KxN, row-major) * scale + bias[m] + resid
__global__ void gemm_kn_kernel(const float* __restrict__ A, long aStr,
                               const float* __restrict__ Bm, long bStr,
                               const float* __restrict__ bias,
                               const float* __restrict__ resid, long rStr,
                               float* __restrict__ Y, long yStr,
                               int M, int N, int K, float scale) {
  const int lane = threadIdx.x;
  const int half = lane >> 4, nl = lane & 15;
  const int n0 = blockIdx.x * 64;
  const int m0 = blockIdx.y * 16;
  const int z  = blockIdx.z;
  const float* Arow = A + (size_t)z * aStr + (size_t)(m0 + nl) * K;
  const float* Bb   = Bm + (size_t)z * bStr;
  v8f acc[4] = {};
  for (int kk = 0; kk < K; kk += 32) {
    v16bf a = load_a_rm(Arow, kk, half);
#pragma unroll
    for (int t = 0; t < 4; ++t) {
      v16bf b = load_b_kn(Bb, N, kk, n0 + t * 16 + nl, half);
      acc[t] = wmma_bf16(a, b, acc[t]);
    }
  }
#pragma unroll
  for (int t = 0; t < 4; ++t) {
    int n = n0 + t * 16 + nl;
#pragma unroll
    for (int r = 0; r < 8; ++r) {
      int m = m0 + half * 8 + r;
      float v = acc[t][r] * scale;
      if (bias)  v += bias[m];
      if (resid) v += resid[(size_t)z * rStr + (size_t)m * N + n];
      Y[(size_t)z * yStr + (size_t)m * N + n] = v;
    }
  }
}

// Y[z] = softA(A[z]) (MxK) @ B[z](NxK)^T * scale ; optional row softmax on A
__global__ void gemm_abt_kernel(const float* __restrict__ A, long aStr,
                                const float* __restrict__ Bm, long bStr,
                                const float* __restrict__ mx,
                                const float* __restrict__ rinv,
                                float* __restrict__ Y, long yStr,
                                int M, int N, int K, float scale) {
  const int lane = threadIdx.x;
  const int half = lane >> 4, nl = lane & 15;
  const int n0 = blockIdx.x * 64;
  const int m0 = blockIdx.y * 16;
  const int z  = blockIdx.z;
  const float* Arow = A + (size_t)z * aStr + (size_t)(m0 + nl) * K;
  const float* Bb   = Bm + (size_t)z * bStr;
  float rMx = 0.f, rRs = 1.f;
  const bool soft = (mx != nullptr);
  if (soft) {
    rMx = mx[(size_t)z * M + m0 + nl];
    rRs = rinv[(size_t)z * M + m0 + nl];
  }
  v8f acc[4] = {};
  for (int kk = 0; kk < K; kk += 32) {
    v16bf a = soft ? load_a_rm_soft(Arow, kk, half, rMx, rRs)
                   : load_a_rm(Arow, kk, half);
#pragma unroll
    for (int t = 0; t < 4; ++t) {
      const float* brow = Bb + (size_t)(n0 + t * 16 + nl) * K;
      v16bf b = load_b_nk(brow, kk, half);
      acc[t] = wmma_bf16(a, b, acc[t]);
    }
  }
#pragma unroll
  for (int t = 0; t < 4; ++t) {
    int n = n0 + t * 16 + nl;
#pragma unroll
    for (int r = 0; r < 8; ++r) {
      int m = m0 + half * 8 + r;
      Y[(size_t)z * yStr + (size_t)m * N + n] = acc[t][r] * scale;
    }
  }
}

// ---------------------------------------------------------------------------
// KMeans kernels
// ---------------------------------------------------------------------------

__global__ void p2_kernel(const float* __restrict__ xt, float* __restrict__ p2) {
  int n = blockIdx.x * 256 + threadIdx.x;
  int b = blockIdx.y;
  const float* X = xt + (size_t)b * CO * HW + n;
  float s = 0.f;
  for (int c = 0; c < CO; ++c) { float v = X[(size_t)c * HW]; s += v * v; }
  p2[(size_t)b * HW + n] = s;
}

// init centroids at n = 65*j (round(linspace(0,4095,64))), and c2
__global__ void kmeans_init_kernel(const float* __restrict__ xt,
                                   float* __restrict__ cent,
                                   float* __restrict__ c2) {
  int j = threadIdx.x;   // 64
  int b = blockIdx.x;    // 8
  int n = 65 * j;
  const float* X = xt + (size_t)b * CO * HW + n;
  float s = 0.f;
  for (int c = 0; c < CO; ++c) {
    float v = X[(size_t)c * HW];
    cent[((size_t)b * KC + j) * CO + c] = v;
    s += v * v;
  }
  c2[(size_t)b * KC + j] = s;
}

// Assignment: per wave, 16 points vs 64 centroids via 16 WMMAs + shfl argmin.
__global__ void kmeans_assign_kernel(const float* __restrict__ xt,
                                     const float* __restrict__ cent,
                                     const float* __restrict__ c2,
                                     const float* __restrict__ p2,
                                     int* __restrict__ labels) {
  const int lane = threadIdx.x & 31;
  const int wave = threadIdx.x >> 5;
  const int half = lane >> 4, nl = lane & 15;
  const int b = blockIdx.y;
  const int n0 = (blockIdx.x * 8 + wave) * 16;
  const float* X  = xt + (size_t)b * CO * HW;
  const float* Cb = cent + (size_t)b * KC * CO;
  v8f acc[4] = {};
  for (int kk = 0; kk < CO; kk += 32) {
    const float* ap = X + (size_t)(kk + half * 16) * HW + n0 + nl;
    v16bf a;
#pragma unroll
    for (int i = 0; i < 8; ++i) {
      int kb = i << 1;
      a[2 * i]     = (__bf16)ap[(size_t)kb * HW];
      a[2 * i + 1] = (__bf16)ap[(size_t)(kb + 1) * HW];
    }
#pragma unroll
    for (int g = 0; g < 4; ++g) {
      const float* crow = Cb + (size_t)(g * 16 + nl) * CO;
      v16bf bfrag = load_b_nk(crow, kk, half);
      acc[g] = wmma_bf16(a, bfrag, acc[g]);
    }
  }
  float c2v[4];
#pragma unroll
  for (int g = 0; g < 4; ++g) c2v[g] = c2[(size_t)b * KC + g * 16 + nl];
  const int base = n0 + half * 8;
#pragma unroll
  for (int r = 0; r < 8; ++r) {
    float pv = p2[(size_t)b * HW + base + r];
    float best = 3.4e38f; int bidx = 0;
#pragma unroll
    for (int g = 0; g < 4; ++g) {
      float d = pv + c2v[g] - 2.f * acc[g][r];
      if (d < best) { best = d; bidx = g * 16 + nl; }
    }
    for (int off = 8; off; off >>= 1) {
      float ov = __shfl_xor(best, off, 16);
      int   oi = __shfl_xor(bidx, off, 16);
      if (ov < best || (ov == best && oi < bidx)) { best = ov; bidx = oi; }
    }
    if (nl == 0) labels[(size_t)b * HW + base + r] = bidx;
  }
}

// Cluster sums/counts via LDS atomics then global-atomic flush.
__global__ void kmeans_accum_kernel(const float* __restrict__ xt,
                                    const int* __restrict__ labels,
                                    float* __restrict__ sums,
                                    float* __restrict__ counts) {
  __shared__ float ls[KC * CO];
  __shared__ float lc[KC];
  const int t = threadIdx.x;
  const int b = blockIdx.y;
  for (int i = t; i < KC * CO; i += 256) ls[i] = 0.f;
  if (t < KC) lc[t] = 0.f;
  __syncthreads();
  const int n = blockIdx.x * 256 + t;
  const int l = labels[(size_t)b * HW + n];
  atomicAdd(&lc[l], 1.0f);
  const float* X = xt + (size_t)b * CO * HW + n;
  for (int c = 0; c < CO; ++c) atomicAdd(&ls[l * CO + c], X[(size_t)c * HW]);
  __syncthreads();
  for (int i = t; i < KC * CO; i += 256)
    if (ls[i] != 0.f) atomicAdd(&sums[(size_t)b * KC * CO + i], ls[i]);
  if (t < KC && lc[t] != 0.f) atomicAdd(&counts[(size_t)b * KC + t], lc[t]);
}

// cent = cnt>0 ? sums/cnt : cent ; recompute c2
__global__ void kmeans_update_kernel(const float* __restrict__ sums,
                                     const float* __restrict__ counts,
                                     float* __restrict__ cent,
                                     float* __restrict__ c2) {
  int j = threadIdx.x;  // 64
  int b = blockIdx.x;   // 8
  float cnt = counts[(size_t)b * KC + j];
  float inv = (cnt > 0.f) ? 1.f / cnt : 0.f;
  float s2 = 0.f;
  for (int c = 0; c < CO; ++c) {
    size_t idx = ((size_t)b * KC + j) * CO + c;
    float v = (cnt > 0.f) ? sums[idx] * inv : cent[idx];
    cent[idx] = v;
    s2 += v * v;
  }
  c2[(size_t)b * KC + j] = s2;
}

// ---------------------------------------------------------------------------
// Softmax / misc kernels
// ---------------------------------------------------------------------------

__global__ void slot_softmax_kernel(const float* __restrict__ st,
                                    float* __restrict__ outm) {
  int n = blockIdx.x * 256 + threadIdx.x;
  int b = blockIdx.y;
  const float* p = st + (size_t)b * SN * HW + n;
  float v[SN];
  float mx = -3.4e38f;
#pragma unroll
  for (int s = 0; s < SN; ++s) { v[s] = p[(size_t)s * HW]; mx = fmaxf(mx, v[s]); }
  float sum = 0.f;
#pragma unroll
  for (int s = 0; s < SN; ++s) { v[s] = __expf(v[s] - mx); sum += v[s]; }
  float r = 1.f / sum;
  float* o = outm + (size_t)b * SN * HW + n;
#pragma unroll
  for (int s = 0; s < SN; ++s) o[(size_t)s * HW] = v[s] * r;
}

__global__ void rowstats_kernel(const float* __restrict__ X,
                                float* __restrict__ mx,
                                float* __restrict__ rinv, int ncols) {
  const int lane = threadIdx.x & 31;
  const int wave = threadIdx.x >> 5;
  const long row = (long)blockIdx.x * 8 + wave;
  const float* p = X + row * (long)ncols;
  float m = -3.4e38f;
  for (int i = lane; i < ncols; i += 32) m = fmaxf(m, p[i]);
  for (int off = 16; off; off >>= 1) m = fmaxf(m, __shfl_xor(m, off, 32));
  float s = 0.f;
  for (int i = lane; i < ncols; i += 32) s += __expf(p[i] - m);
  for (int off = 16; off; off >>= 1) s += __shfl_xor(s, off, 32);
  if (lane == 0) { mx[row] = m; rinv[row] = 1.f / s; }
}

__global__ void softmax_rows_kernel(float* __restrict__ X, int ncols) {
  const int lane = threadIdx.x & 31;
  const int wave = threadIdx.x >> 5;
  const long row = (long)blockIdx.x * 8 + wave;
  float* p = X + row * (long)ncols;
  float m = -3.4e38f;
  for (int i = lane; i < ncols; i += 32) m = fmaxf(m, p[i]);
  for (int off = 16; off; off >>= 1) m = fmaxf(m, __shfl_xor(m, off, 32));
  float s = 0.f;
  for (int i = lane; i < ncols; i += 32) s += __expf(p[i] - m);
  for (int off = 16; off; off >>= 1) s += __shfl_xor(s, off, 32);
  float r = 1.f / s;
  for (int i = lane; i < ncols; i += 32) p[i] = __expf(p[i] - m) * r;
}

// kv[b][r*CO+i][j] = ref_pt[r][b][i][j]
__global__ void kv_copy_kernel(const float* __restrict__ refpt,
                               float* __restrict__ kv) {
  long i = (long)blockIdx.x * 256 + threadIdx.x;
  int j  = (int)(i & (CO - 1));
  long t = i >> 7;
  int rc = (int)(t % KV);
  int b  = (int)(t / KV);
  int r  = rc >> 7;
  int ii = rc & (CO - 1);
  kv[i] = refpt[(((size_t)(r * BN + b) * CO + ii) * CO) + j];
}

__global__ void zero_kernel(float* __restrict__ p, long n) {
  long i = (long)blockIdx.x * 256 + threadIdx.x;
  for (; i < n; i += (long)gridDim.x * 256) p[i] = 0.f;
}

// ---------------------------------------------------------------------------
// Host orchestration
// ---------------------------------------------------------------------------
extern "C" void kernel_launch(void* const* d_in, const int* in_sizes, int n_in,
                              void* d_out, int out_size, void* d_ws, size_t ws_size,
                              hipStream_t stream) {
  const float* target = (const float*)d_in[0];
  const float* ref    = (const float*)d_in[1];
  const float* sqz_w  = (const float*)d_in[2];
  const float* sqz_b  = (const float*)d_in[3];
  const float* slot_w = (const float*)d_in[4];
  const float* slot_b = (const float*)d_in[5];
  const float* Wq     = (const float*)d_in[6];
  const float* Wk     = (const float*)d_in[7];
  const float* Wv     = (const float*)d_in[8];
  const float* Wo     = (const float*)d_in[9];

  float* out = (float*)d_out;
  float* o_sqz   = out;
  float* o_agg   = out + (size_t)BN * CO * HW;
  float* o_refpt = o_agg + (size_t)BN * KC * CO;
  float* o_slpt  = o_refpt + (size_t)RB * CO * CO;
  float* o_slmsk = o_slpt + (size_t)BN * SN * CO;

  float* ws = (float*)d_ws;
  float* w_slot   = ws;                                size_t off = (size_t)BN * SN * HW;
  float* w_xr     = ws + off;                          off += (size_t)RB * CO * HW;
  float* w_p2     = ws + off;                          off += (size_t)BN * HW;
  float* w_cent   = ws + off;                          off += (size_t)BN * KC * CO;
  float* w_c2     = ws + off;                          off += (size_t)BN * KC;
  float* w_sums   = ws + off;                          off += (size_t)BN * KC * CO;
  float* w_counts = ws + off;                          off += (size_t)BN * KC;
  int*   w_labels = (int*)(ws + off);                  off += (size_t)BN * HW;
  float* w_tpt    = ws + off;                          off += (size_t)BN * KC * CO;
  float* w_mx     = ws + off;                          off += (size_t)RB * CO;
  float* w_rinv   = ws + off;                          off += (size_t)RB * CO;
  float* w_kv     = ws + off;                          off += (size_t)BN * KV * CO;
  float* w_q      = ws + off;                          off += (size_t)BN * KC * CO;
  float* w_k      = ws + off;                          off += (size_t)BN * KV * CO;
  float* w_v      = ws + off;                          off += (size_t)BN * KV * CO;
  float* w_attn   = ws + off;                          off += (size_t)BN * KC * KV;
  float* w_av     = ws + off;                          off += (size_t)BN * KC * CO;

  const dim3 wv(32);

  // 1) sqz_target = sqz_w @ target + b  -> out region 0 (this is xt)
  conv_gemm_kernel<<<dim3(HW / 128, 1, BN), 256, 0, stream>>>(
      sqz_w, sqz_b, target, (long)CIN * HW, o_sqz, (long)CO * HW, CIN);
  // 2) slot_target -> ws (M=16, small: generic path)
  gemm_kn_kernel<<<dim3(HW / 64, SN / 16, BN), wv, 0, stream>>>(
      slot_w, 0, target, (long)CIN * HW, slot_b, nullptr, 0,
      w_slot, (long)SN * HW, SN, HW, CIN, 1.f);
  // 3) xr = sqz over refs -> ws (24 batches)
  conv_gemm_kernel<<<dim3(HW / 128, 1, RB), 256, 0, stream>>>(
      sqz_w, sqz_b, ref, (long)CIN * HW, w_xr, (long)CO * HW, CIN);

  // 4) point norms, 5) centroid init
  p2_kernel<<<dim3(HW / 256, BN), 256, 0, stream>>>(o_sqz, w_p2);
  kmeans_init_kernel<<<BN, KC, 0, stream>>>(o_sqz, w_cent, w_c2);

  // 6) KMeans iterations
  for (int it = 0; it < KM_ITERS; ++it) {
    kmeans_assign_kernel<<<dim3(32, BN), 256, 0, stream>>>(
        o_sqz, w_cent, w_c2, w_p2, w_labels);
    zero_kernel<<<128, 256, 0, stream>>>(w_sums,
        (long)BN * KC * CO + (long)BN * KC);
    kmeans_accum_kernel<<<dim3(HW / 256, BN), 256, 0, stream>>>(
        o_sqz, w_labels, w_sums, w_counts);
    kmeans_update_kernel<<<BN, KC, 0, stream>>>(w_sums, w_counts, w_cent, w_c2);
  }
  // 7) final assignment + target_pt = per-cluster sums
  kmeans_assign_kernel<<<dim3(32, BN), 256, 0, stream>>>(
      o_sqz, w_cent, w_c2, w_p2, w_labels);
  zero_kernel<<<128, 256, 0, stream>>>(w_tpt, (long)BN * KC * CO);
  zero_kernel<<<2, 256, 0, stream>>>(w_counts, (long)BN * KC);
  kmeans_accum_kernel<<<dim3(HW / 256, BN), 256, 0, stream>>>(
      o_sqz, w_labels, w_tpt, w_counts);

  // 8) slot softmax -> slot_mask, then slot_pt = ss @ xt^T
  slot_softmax_kernel<<<dim3(HW / 256, BN), 256, 0, stream>>>(w_slot, o_slmsk);
  gemm_abt_kernel<<<dim3(CO / 64, SN / 16, BN), wv, 0, stream>>>(
      o_slmsk, (long)SN * HW, o_sqz, (long)CO * HW, nullptr, nullptr,
      o_slpt, (long)SN * CO, SN, CO, HW, 1.f);

  // 9) ref prototypes: row softmax stats, then ref_pt = ssr @ xr^T (fused)
  rowstats_kernel<<<(RB * CO) / 8, 256, 0, stream>>>(w_xr, w_mx, w_rinv, HW);
  gemm_abt_kernel<<<dim3(CO / 64, CO / 16, RB), wv, 0, stream>>>(
      w_xr, (long)CO * HW, w_xr, (long)CO * HW, w_mx, w_rinv,
      o_refpt, (long)CO * CO, CO, CO, HW, 1.f);

  // 10) attention
  kv_copy_kernel<<<((long)BN * KV * CO) / 256, 256, 0, stream>>>(o_refpt, w_kv);
  gemm_kn_kernel<<<dim3(CO / 64, KC / 16, BN), wv, 0, stream>>>(
      w_tpt, (long)KC * CO, Wq, 0, nullptr, nullptr, 0,
      w_q, (long)KC * CO, KC, CO, CO, 1.f);
  gemm_kn_kernel<<<dim3(CO / 64, KV / 16, BN), wv, 0, stream>>>(
      w_kv, (long)KV * CO, Wk, 0, nullptr, nullptr, 0,
      w_k, (long)KV * CO, KV, CO, CO, 1.f);
  gemm_kn_kernel<<<dim3(CO / 64, KV / 16, BN), wv, 0, stream>>>(
      w_kv, (long)KV * CO, Wv, 0, nullptr, nullptr, 0,
      w_v, (long)KV * CO, KV, CO, CO, 1.f);
  gemm_abt_kernel<<<dim3(KV / 64, KC / 16, BN), wv, 0, stream>>>(
      w_q, (long)KC * CO, w_k, (long)KV * CO, nullptr, nullptr,
      w_attn, (long)KC * KV, KC, KV, CO, 0.08838834764831845f);
  softmax_rows_kernel<<<(BN * KC) / 8, 256, 0, stream>>>(w_attn, KV);
  gemm_kn_kernel<<<dim3(CO / 64, KC / 16, BN), wv, 0, stream>>>(
      w_attn, (long)KC * KV, w_v, (long)KV * CO, nullptr, nullptr, 0,
      w_av, (long)KC * CO, KC, CO, KV, 1.f);
  gemm_kn_kernel<<<dim3(CO / 64, KC / 16, BN), wv, 0, stream>>>(
      w_av, (long)KC * CO, Wo, 0, nullptr, w_tpt, (long)KC * CO,
      o_agg, (long)KC * CO, KC, CO, CO, 1.f);

  (void)in_sizes; (void)n_in; (void)out_size; (void)ws_size;
}
